// L2ContrastiveLoss_7868380086883
// MI455X (gfx1250) — matrix operand order repaired and checked by reading, not compiled, not run on hardware
//
#include <hip/hip_runtime.h>
#include <cstdint>
#include <cstddef>

// N=8192 rows, D=1024 features, margin=1, eps=1e-12
#define N_ROWS 8192
#define DIMK   1024
#define TILE   128    // output tile is TILE x TILE
#define KST    32     // K-step = bf16 WMMA K
#define LROWH  40     // padded LDS row stride in bf16 (80B: 16B-aligned, bank-spread)
#define NKS    (DIMK / KST)

typedef __attribute__((ext_vector_type(16))) __bf16 v16bf;
typedef __attribute__((ext_vector_type(8)))  __bf16 v8bf;
typedef __attribute__((ext_vector_type(4)))  __bf16 v4bf;
typedef __attribute__((ext_vector_type(8)))  float  v8f;
typedef __attribute__((ext_vector_type(4)))  int    v4i;

// Pointer types matching the async-to-LDS builtin signature (vV4i*1 V4i*3 Ii Ii)
typedef v4i __attribute__((address_space(1)))* global_v4i_ptr;
typedef v4i __attribute__((address_space(3)))* lds_v4i_ptr;

#if __has_builtin(__builtin_amdgcn_global_load_async_to_lds_b128) && \
    __has_builtin(__builtin_amdgcn_s_wait_asynccnt)
#define USE_ASYNC 1
#else
#define USE_ASYNC 0
#endif

// Copy 16 bytes global -> LDS via the CDNA5 async path (ASYNCcnt-tracked).
__device__ __forceinline__ void copy16(const __bf16* __restrict__ g, __bf16* l) {
#if USE_ASYNC
  __builtin_amdgcn_global_load_async_to_lds_b128(
      (global_v4i_ptr)(g), (lds_v4i_ptr)(l), 0, 0);
#else
  *(v4i*)l = *(const v4i*)g;
#endif
}

union frag16 { v16bf v; v8bf h[2]; };

// ------- Kernel 1: fp32 -> bf16 copies + row norms + accumulator reset -------
// One pass over the 64 MiB inputs; bf16 working set (32 MiB) then lives in L2.
__global__ __launch_bounds__(256) void l2cl_prep(const float* __restrict__ f1,
                                                 const float* __restrict__ f2,
                                                 __bf16* __restrict__ g1b,
                                                 __bf16* __restrict__ g2b,
                                                 float* __restrict__ sq,
                                                 double* __restrict__ acc) {
  const int b = blockIdx.x;            // [0, 2N): first N -> f1 rows, rest -> f2 rows
  const int t = threadIdx.x;           // 256 threads * 4 floats = 1024 = D
  const bool first = (b < N_ROWS);
  const int row = first ? b : (b - N_ROWS);
  const float*  src = (first ? f1 : f2) + (size_t)row * DIMK;
  __bf16*       dst = (first ? g1b : g2b) + (size_t)row * DIMK;

  const float4 v = ((const float4*)src)[t];
  v4bf o;
  o[0] = (__bf16)v.x; o[1] = (__bf16)v.y; o[2] = (__bf16)v.z; o[3] = (__bf16)v.w;
  *(v4bf*)(dst + 4 * t) = o;

  float s = v.x * v.x + v.y * v.y + v.z * v.z + v.w * v.w;
#pragma unroll
  for (int ofs = 16; ofs > 0; ofs >>= 1) s += __shfl_down(s, ofs, 32);
  __shared__ float red[8];
  if ((t & 31) == 0) red[t >> 5] = s;
  __syncthreads();
  if (t == 0) {
    float tot = 0.f;
#pragma unroll
    for (int w = 0; w < 8; ++w) tot += red[w];
    sq[b] = tot;
    if (b == 0) *acc = 0.0;            // re-zero every call (graph replay safe)
  }
}

// ------- Kernel 2: fused WMMA tile GEMM + loss reduction (double-buffered) -------
__global__ __launch_bounds__(256) void l2cl_tile(const __bf16* __restrict__ g1b,
                                                 const __bf16* __restrict__ g2b,
                                                 const float* __restrict__ sq,
                                                 double* __restrict__ acc) {
  __shared__ __bf16 ldsA[2][TILE * LROWH];   // 2 x 10 KB
  __shared__ __bf16 ldsB[2][TILE * LROWH];   // 2 x 10 KB

  const int t    = threadIdx.x;
  const int lane = t & 31;
  const int wave = t >> 5;               // 8 waves
  const int bi = blockIdx.y * TILE;      // f1 row base
  const int bj = blockIdx.x * TILE;      // f2 row base (= output col base)
  const int wm = (wave & 3) * 32;        // wave sub-tile: 32 rows
  const int wn = (wave >> 2) * 64;       //                64 cols

  const v8f vzero = {0.f, 0.f, 0.f, 0.f, 0.f, 0.f, 0.f, 0.f};
  v8f accv[2][4];
#pragma unroll
  for (int mi = 0; mi < 2; ++mi)
#pragma unroll
    for (int ni = 0; ni < 4; ++ni) accv[mi][ni] = vzero;

  const int l15 = lane & 15;
  const int hiA = (lane & 16) ? 8 : 0;   // A frag: hi half-wave K pattern +8
  const int hiB = (lane & 16) ? 16 : 0;  // unused; B frags load with hiA (same K order as A)

  // Stage one 128x32 bf16 tile of each matrix: 128 rows x 4 segs of 16B = 512
  // transfers, 2 per thread per matrix.
  auto stage = [&](int buf, int kofs) {
#pragma unroll
    for (int it = 0; it < 2; ++it) {
      const int id  = it * 256 + t;      // [0,512)
      const int row = id >> 2;
      const int seg = id & 3;            // 8 bf16 = 16B per seg
      copy16(g1b + (size_t)(bi + row) * DIMK + kofs + seg * 8,
             &ldsA[buf][row * LROWH + seg * 8]);
      copy16(g2b + (size_t)(bj + row) * DIMK + kofs + seg * 8,
             &ldsB[buf][row * LROWH + seg * 8]);
    }
  };

  int cur = 0;
  stage(0, 0);
#if USE_ASYNC
  __builtin_amdgcn_s_wait_asynccnt(0);
#endif
  __syncthreads();

  for (int ks = 0; ks < NKS; ++ks) {
    if (ks + 1 < NKS) stage(cur ^ 1, (ks + 1) * KST);   // prefetch next K-step

    // ---- fragments: two 16B ds_load_b128 each, per CDNA5 16-bit layouts ----
    v16bf afr[2], bfr[4];
#pragma unroll
    for (int mi = 0; mi < 2; ++mi) {
      const __bf16* rp = &ldsA[cur][(wm + mi * 16 + l15) * LROWH];
      frag16 f;
      f.h[0] = *(const v8bf*)(rp + hiA);        // K {0..7}  (+8 for hi half)
      f.h[1] = *(const v8bf*)(rp + hiA + 16);   // K {16..23}(+8 for hi half)
      afr[mi] = f.v;
    }
#pragma unroll
    for (int ni = 0; ni < 4; ++ni) {
      const __bf16* rp = &ldsB[cur][(wn + ni * 16 + l15) * LROWH];
      frag16 f;
      f.h[0] = *(const v8bf*)(rp + hiA);        // K {0..7}  (+8 for hi half)
      f.h[1] = *(const v8bf*)(rp + hiA + 16);   // K {16..23}(+8 for hi half)
      bfr[ni] = f.v;
    }

    // ---- 8 WMMAs per wave per K-step ----
#pragma unroll
    for (int mi = 0; mi < 2; ++mi)
#pragma unroll
      for (int ni = 0; ni < 4; ++ni)
        accv[mi][ni] = __builtin_amdgcn_wmma_f32_16x16x32_bf16(
            false, afr[mi], false, bfr[ni], (short)0, accv[mi][ni], false, false);

#if USE_ASYNC
    __builtin_amdgcn_s_wait_asynccnt(0);
#endif
    __syncthreads();
    cur ^= 1;
  }

  // ---- fused loss epilogue on accumulators ----
  // C/D layout: VGPR r, lanes 0-15 -> (M=r, N=lane); lanes 16-31 -> (M=r+8, N=lane-16)
  float lsum = 0.f;
#pragma unroll
  for (int mi = 0; mi < 2; ++mi) {
    const int i0 = bi + wm + mi * 16 + hiA;
#pragma unroll
    for (int ni = 0; ni < 4; ++ni) {
      const int j = bj + wn + ni * 16 + l15;
      const float s2 = sq[N_ROWS + j];
#pragma unroll
      for (int r = 0; r < 8; ++r) {
        const int i = i0 + r;
        float d2 = sq[i] + s2 - 2.0f * accv[mi][ni][r];
        d2 = fmaxf(d2, 0.0f);
        const float s = sqrtf(d2 + 1e-12f);
        const float h = fmaxf(1.0f - s, 0.0f);
        lsum += (i == j) ? d2 : h * h;
      }
    }
  }
#pragma unroll
  for (int ofs = 16; ofs > 0; ofs >>= 1) lsum += __shfl_down(lsum, ofs, 32);
  if (lane == 0) atomicAdd(acc, (double)lsum);
}

// ---------------- Kernel 3: finalize ----------------
__global__ void l2cl_finalize(const double* __restrict__ acc, float* __restrict__ out) {
  if (threadIdx.x == 0 && blockIdx.x == 0)
    out[0] = (float)(*acc * (1.0 / (2.0 * (double)N_ROWS)));
}

extern "C" void kernel_launch(void* const* d_in, const int* in_sizes, int n_in,
                              void* d_out, int out_size, void* d_ws, size_t ws_size,
                              hipStream_t stream) {
  (void)in_sizes; (void)n_in; (void)out_size; (void)ws_size;
  const float* f1 = (const float*)d_in[0];
  const float* f2 = (const float*)d_in[1];
  float* out = (float*)d_out;

  // Workspace layout: bf16 f1 (16 MiB) | bf16 f2 (16 MiB) | sq (64 KB) | acc (8 B)
  char* ws = (char*)d_ws;
  const size_t bf_bytes = (size_t)N_ROWS * DIMK * sizeof(__bf16);
  __bf16* g1b = (__bf16*)ws;
  __bf16* g2b = (__bf16*)(ws + bf_bytes);
  float*  sq  = (float*)(ws + 2 * bf_bytes);
  double* acc = (double*)(ws + 2 * bf_bytes + 2 * N_ROWS * sizeof(float));

  l2cl_prep<<<2 * N_ROWS, 256, 0, stream>>>(f1, f2, g1b, g2b, sq, acc);
  dim3 grid(N_ROWS / TILE, N_ROWS / TILE);   // 64 x 64
  l2cl_tile<<<grid, 256, 0, stream>>>(g1b, g2b, sq, acc);
  l2cl_finalize<<<1, 1, 0, stream>>>(acc, out);
}
